// AttentionHead_63720134803563
// MI455X (gfx1250) — matrix operand order repaired and checked by reading, not compiled, not run on hardware
//
#include <hip/hip_runtime.h>
#include <hip/hip_bf16.h>
#include <math.h>

// ---------------------------------------------------------------------------
// Causal single-head attention for gfx1250 (MI455X, wave32, WMMA).
//   S = 4096, D_MODEL = 1024, D_K = 64, scale = 1/sqrt(64) = 0.125
// Pipeline:
//   0) cvt_f32_to_f16_kernel : one-time convert of x, Wq, Wk, Wv to f16
//   1) qkv_proj_kernel       : Q/K = x @ W.T + b (f16, Q pre-scaled),
//                              V stored TRANSPOSED as VT[64][4096] f16
//   2) attn_kernel           : flash-attention causal softmax(QK^T)V,
//                              32 keys per iteration (full K=32 WMMAs)
// All matmuls use v_wmma_f32_16x16x32_f16 with fp32 accumulation.
// ---------------------------------------------------------------------------

#define S_DIM   4096
#define D_MODEL 1024
#define D_K     64

typedef __attribute__((ext_vector_type(16))) _Float16 v16h;
typedef __attribute__((ext_vector_type(8)))  _Float16 v8h;
typedef __attribute__((ext_vector_type(8)))  float    v8f;
typedef __attribute__((ext_vector_type(4)))  float    v4f;

// ---------------------------------------------------------------------------
// Kernel 0: vectorized f32 -> f16 conversion (8 elements / thread).
// ---------------------------------------------------------------------------
__global__ __launch_bounds__(256) void cvt_f32_to_f16_kernel(
    const float* __restrict__ src, _Float16* __restrict__ dst, int n8)
{
    int i = blockIdx.x * blockDim.x + threadIdx.x;
    if (i >= n8) return;
    const float* p = src + (size_t)i * 8;
    v4f a = *(const v4f*)(p);
    v4f b = *(const v4f*)(p + 4);
    v8h h;
    #pragma unroll
    for (int j = 0; j < 4; ++j) {
        h[j]     = (_Float16)a[j];
        h[4 + j] = (_Float16)b[j];
    }
    *(v8h*)(dst + (size_t)i * 8) = h;
}

// ---------------------------------------------------------------------------
// Kernel 1: fused QKV projection (all-f16 operands, pure b128 loads).
//
// 16-bit A (16x32) per-lane layout (ISA 7.12.2):
//   lane L (0..15): row M=L,      halves 0..7 -> K=0..7,  8..15 -> K=16..23
//   lane L (16..31): row M=L-16,  halves 0..7 -> K=8..15, 8..15 -> K=24..31
// 16-bit B (32x16): lanes 0..15 col N=lane, halves 0..7 -> K=0..7, 8..15 -> K=16..23;
//                   lanes 16..31 col N=lane-16, halves 0..7 -> K=8..15, 8..15 -> K=24..31
// f32 C/D (16x16): VGPR r: lanes 0..15 (M=r, N=lane); 16..31 (M=r+8, N=lane-16)
// ---------------------------------------------------------------------------
__global__ __launch_bounds__(128) void qkv_proj_kernel(
    const _Float16* __restrict__ x16,
    const _Float16* __restrict__ wq16, const float* __restrict__ bq,
    const _Float16* __restrict__ wk16, const float* __restrict__ bk,
    const _Float16* __restrict__ wv16, const float* __restrict__ bv,
    _Float16* __restrict__ q16, _Float16* __restrict__ k16,
    _Float16* __restrict__ vt16)
{
    const int lane = threadIdx.x & 31;
    const int wave = threadIdx.x >> 5;
    const int tile = blockIdx.x * 4 + wave;   // 16-row tile index, 0..255
    const int hi   = lane >> 4;
    const int ln   = lane & 15;

    const _Float16* Wm[3] = { wq16, wk16, wv16 };

    v8f zero8f = {};
    v8f acc[3][4];
    #pragma unroll
    for (int m = 0; m < 3; ++m)
        #pragma unroll
        for (int nt = 0; nt < 4; ++nt)
            acc[m][nt] = zero8f;

    const int rowA = tile * 16 + ln;

    for (int kt = 0; kt < D_MODEL / 32; ++kt) {
        // ---- A tile: x16[rowA][kt*32 ..], already in A layout chunks ----
        const _Float16* xr = x16 + (size_t)rowA * D_MODEL + kt * 32 + hi * 8;
        v8h alo = *(const v8h*)(xr);
        v8h ahi = *(const v8h*)(xr + 16);
        v16h a = __builtin_shufflevector(alo, ahi,
            0, 1, 2, 3, 4, 5, 6, 7, 8, 9, 10, 11, 12, 13, 14, 15);

        // ---- B tiles: W16 is [64][1024]; B[c][n] = W[n][c] ----
        #pragma unroll
        for (int m = 0; m < 3; ++m) {
            #pragma unroll
            for (int nt = 0; nt < 4; ++nt) {
                const int ncol = nt * 16 + ln;
                const _Float16* wr = Wm[m] + (size_t)ncol * D_MODEL + kt * 32 + hi * 8;
                v8h blo = *(const v8h*)(wr);
                v8h bhi = *(const v8h*)(wr + 16);
                v16h b = __builtin_shufflevector(blo, bhi,
                    0, 1, 2, 3, 4, 5, 6, 7, 8, 9, 10, 11, 12, 13, 14, 15);
                acc[m][nt] = __builtin_amdgcn_wmma_f32_16x16x32_f16(
                    false, a, false, b, (short)0, acc[m][nt], false, false);
            }
        }
    }

    // ---- Q/K: bias add (+ Q scale), row-major f16 stores ----
    #pragma unroll
    for (int nt = 0; nt < 4; ++nt) {
        const int col = nt * 16 + ln;
        const float bqv = bq[col];
        const float bkv = bk[col];
        #pragma unroll
        for (int r = 0; r < 8; ++r) {
            const int row = tile * 16 + r + 8 * hi;
            q16[(size_t)row * D_K + col] = (_Float16)((acc[0][nt][r] + bqv) * 0.125f);
            k16[(size_t)row * D_K + col] = (_Float16)(acc[1][nt][r] + bkv);
        }
    }
    // ---- V: bias add, store transposed VT[col][row] (contiguous v8h) ----
    #pragma unroll
    for (int nt = 0; nt < 4; ++nt) {
        const int col = nt * 16 + ln;
        const float bvv = bv[col];
        v8h pack;
        #pragma unroll
        for (int r = 0; r < 8; ++r)
            pack[r] = (_Float16)(acc[2][nt][r] + bvv);
        *(v8h*)(vt16 + (size_t)col * S_DIM + tile * 16 + 8 * hi) = pack;
    }
}

// ---------------------------------------------------------------------------
// Kernel 2: causal flash attention, 32 keys per iteration.
//   S(16x32) = Q @ K^T              (4 WMMAs: 2 subtiles x K=64)
//   online softmax over 32 keys     (16-lane __shfl_xor reductions)
//   O += P(16x32) @ V(32x64)        (4 WMMAs, full K=32, B from VT)
// P transposed C-layout -> A-layout through per-wave LDS (16x32 halfs).
// ---------------------------------------------------------------------------
__global__ __launch_bounds__(256) void attn_kernel(
    const _Float16* __restrict__ q16,
    const _Float16* __restrict__ k16,
    const _Float16* __restrict__ vt16,
    float* __restrict__ out)
{
    __shared__ _Float16 lds_p[8][16 * 32];   // per-wave 16x32 P staging (8 KB)

    const int lane = threadIdx.x & 31;
    const int wave = threadIdx.x >> 5;
    // spread short/long causal tiles evenly across blocks
    const int qi   = wave * gridDim.x + blockIdx.x;   // query tile, 0..255
    const int hi   = lane >> 4;
    const int ln   = lane & 15;

    // ---- preload Q tile as two 16x32 f16 A-operands ----
    v16h qa[2];
    {
        const int row = qi * 16 + ln;
        #pragma unroll
        for (int c = 0; c < 2; ++c) {
            const _Float16* p = q16 + (size_t)row * D_K + c * 32 + hi * 8;
            v8h lo = *(const v8h*)(p);
            v8h hc = *(const v8h*)(p + 16);
            qa[c] = __builtin_shufflevector(lo, hc,
                0, 1, 2, 3, 4, 5, 6, 7, 8, 9, 10, 11, 12, 13, 14, 15);
        }
    }

    v8f zero8f = {};
    float mrow[8], lrow[8];
    v8f o[4];
    #pragma unroll
    for (int r = 0; r < 8; ++r) { mrow[r] = -INFINITY; lrow[r] = 0.f; }
    #pragma unroll
    for (int nt = 0; nt < 4; ++nt) o[nt] = zero8f;

    const int nb = (qi >> 1) + 1;            // 32-key blocks (causal)
    for (int b = 0; b < nb; ++b) {
        const int kbase = b * 32;
        // prefetch next key block rows (global_prefetch_b8)
        __builtin_prefetch(k16 + (size_t)(kbase + 32 + ln) * D_K, 0, 1);

        // ---- S subtiles: S[t] = Q @ K^T for keys kbase+t*16.. ----
        v8f s[2];
        #pragma unroll
        for (int t = 0; t < 2; ++t) {
            s[t] = zero8f;
            #pragma unroll
            for (int c = 0; c < 2; ++c) {
                const _Float16* p = k16 +
                    (size_t)(kbase + t * 16 + ln) * D_K + c * 32 + hi * 8;
                v8h lo = *(const v8h*)(p);
                v8h hc = *(const v8h*)(p + 16);
                v16h kb = __builtin_shufflevector(lo, hc,
                    0, 1, 2, 3, 4, 5, 6, 7, 8, 9, 10, 11, 12, 13, 14, 15);
                s[t] = __builtin_amdgcn_wmma_f32_16x16x32_f16(
                    false, qa[c], false, kb, (short)0, s[t], false, false);
            }
        }

        // ---- causal mask + online softmax over 32 keys ----
        float pex[2][8], alpha[8];
        #pragma unroll
        for (int r = 0; r < 8; ++r) {
            const int qrow = qi * 16 + r + 8 * hi;
            const int k0 = kbase + ln;
            const int k1 = kbase + 16 + ln;
            float sv0 = (k0 > qrow) ? -INFINITY : s[0][r];
            float sv1 = (k1 > qrow) ? -INFINITY : s[1][r];
            // row max across 32 keys: local pair max, then 16-lane tree
            float mt = fmaxf(sv0, sv1);
            mt = fmaxf(mt, __shfl_xor(mt, 1, 32));
            mt = fmaxf(mt, __shfl_xor(mt, 2, 32));
            mt = fmaxf(mt, __shfl_xor(mt, 4, 32));
            mt = fmaxf(mt, __shfl_xor(mt, 8, 32));
            const float mnew = fmaxf(mrow[r], mt);
            alpha[r] = __expf(mrow[r] - mnew);     // exp(-inf)=0 on first blk
            const float p0 = __expf(sv0 - mnew);   // masked -> 0
            const float p1 = __expf(sv1 - mnew);
            float ls = p0 + p1;
            ls += __shfl_xor(ls, 1, 32);
            ls += __shfl_xor(ls, 2, 32);
            ls += __shfl_xor(ls, 4, 32);
            ls += __shfl_xor(ls, 8, 32);
            lrow[r] = lrow[r] * alpha[r] + ls;
            mrow[r] = mnew;
            pex[0][r] = p0;
            pex[1][r] = p1;
        }

        // ---- P: C-layout -> row-major 16x32 LDS -> A-layout ----
        #pragma unroll
        for (int t = 0; t < 2; ++t)
            #pragma unroll
            for (int r = 0; r < 8; ++r)
                lds_p[wave][(r + 8 * hi) * 32 + t * 16 + ln] = (_Float16)pex[t][r];
        asm volatile("s_wait_dscnt 0" ::: "memory");   // intra-wave LDS RAW
        v8h plo = *(const v8h*)&lds_p[wave][ln * 32 + hi * 8];
        v8h phi = *(const v8h*)&lds_p[wave][ln * 32 + 16 + hi * 8];
        v16h pa = __builtin_shufflevector(plo, phi,
            0, 1, 2, 3, 4, 5, 6, 7, 8, 9, 10, 11, 12, 13, 14, 15);

        // ---- O = O*alpha + P @ V  (B[k][n] = VT[n][kbase+k], contiguous) ----
        #pragma unroll
        for (int nt = 0; nt < 4; ++nt) {
            v8f ot = o[nt];
            #pragma unroll
            for (int r = 0; r < 8; ++r) ot[r] *= alpha[r];
            const _Float16* vp = vt16 +
                (size_t)(nt * 16 + ln) * S_DIM + kbase + 8 * hi;
            v8h vlo = *(const v8h*)(vp);
            v8h vhi = *(const v8h*)(vp + 16);
            v16h vb = __builtin_shufflevector(vlo, vhi,
                0, 1, 2, 3, 4, 5, 6, 7, 8, 9, 10, 11, 12, 13, 14, 15);
            o[nt] = __builtin_amdgcn_wmma_f32_16x16x32_f16(
                false, pa, false, vb, (short)0, ot, false, false);
        }
    }

    // ---- normalize and write fp32 output ----
    #pragma unroll
    for (int nt = 0; nt < 4; ++nt) {
        #pragma unroll
        for (int r = 0; r < 8; ++r) {
            const int row = qi * 16 + r + 8 * hi;
            out[(size_t)row * D_K + nt * 16 + ln] = o[nt][r] / lrow[r];
        }
    }
}

// ---------------------------------------------------------------------------
extern "C" void kernel_launch(void* const* d_in, const int* in_sizes, int n_in,
                              void* d_out, int out_size, void* d_ws, size_t ws_size,
                              hipStream_t stream)
{
    const float* x  = (const float*)d_in[0];
    const float* Wq = (const float*)d_in[1];
    const float* bq = (const float*)d_in[2];
    const float* Wk = (const float*)d_in[3];
    const float* bk = (const float*)d_in[4];
    const float* Wv = (const float*)d_in[5];
    const float* bv = (const float*)d_in[6];
    float* out = (float*)d_out;

    // workspace layout (f16):
    //   x16  : S*D_MODEL      (8 MB)
    //   wq16 | wk16 | wv16 : 3 * D_K*D_MODEL (384 KB)
    //   q16 | k16 : 2 * S*D_K (1 MB)
    //   vt16 : D_K*S          (512 KB)
    _Float16* x16  = (_Float16*)d_ws;
    _Float16* wq16 = x16  + (size_t)S_DIM * D_MODEL;
    _Float16* wk16 = wq16 + (size_t)D_K * D_MODEL;
    _Float16* wv16 = wk16 + (size_t)D_K * D_MODEL;
    _Float16* q16  = wv16 + (size_t)D_K * D_MODEL;
    _Float16* k16  = q16  + (size_t)S_DIM * D_K;
    _Float16* vt16 = k16  + (size_t)S_DIM * D_K;

    // 0) one-time f32 -> f16 conversions
    const int nx8 = (S_DIM * D_MODEL) / 8;
    const int nw8 = (D_K * D_MODEL) / 8;
    cvt_f32_to_f16_kernel<<<(nx8 + 255) / 256, 256, 0, stream>>>(x,  x16,  nx8);
    cvt_f32_to_f16_kernel<<<(nw8 + 255) / 256, 256, 0, stream>>>(Wq, wq16, nw8);
    cvt_f32_to_f16_kernel<<<(nw8 + 255) / 256, 256, 0, stream>>>(Wk, wk16, nw8);
    cvt_f32_to_f16_kernel<<<(nw8 + 255) / 256, 256, 0, stream>>>(Wv, wv16, nw8);

    // 1) QKV projection: 256 row-tiles, 4 waves/block -> 64 blocks
    qkv_proj_kernel<<<S_DIM / (16 * 4), 128, 0, stream>>>(
        x16, wq16, bq, wk16, bk, wv16, bv, q16, k16, vt16);

    // 2) attention: 256 query tiles, 8 waves/block -> 32 blocks
    attn_kernel<<<S_DIM / (16 * 8), 256, 0, stream>>>(q16, k16, vt16, out);
}
